// GraphResBlock_1211180777898
// MI455X (gfx1250) — matrix operand order repaired and checked by reading, not compiled, not run on hardware
//
#include <hip/hip_runtime.h>

// ---------------------------------------------------------------------------
// GraphResBlock (2x ChebConv K=2 + ReLU + residual) for gfx1250 (MI455X)
//  - bf16 WMMA (v_wmma_f32_16x16x32_bf16) for the 4 GEMMs, f32 accumulate
//  - all f32->bf16 conversion hoisted out of the GEMM (pure b128 loads + WMMA)
//  - edge-parallel f32 atomic scatter for the normalized-Laplacian matvec
// ---------------------------------------------------------------------------

typedef __attribute__((ext_vector_type(16))) __bf16        bf16x16;
typedef __attribute__((ext_vector_type(8)))  float         f32x8;
typedef __attribute__((ext_vector_type(4)))  unsigned int  u32x4;

#define CCH 128  // channels (in == out)

__device__ __forceinline__ unsigned short f2bf(float f) {
    // round-to-nearest-even f32 -> bf16
    unsigned int u = __float_as_uint(f);
    unsigned int r = u + 0x7FFFu + ((u >> 16) & 1u);
    return (unsigned short)(r >> 16);
}
__device__ __forceinline__ unsigned int pk2(float lo, float hi) {
    return (unsigned int)f2bf(lo) | ((unsigned int)f2bf(hi) << 16);
}

union Frag16 { bf16x16 v; u32x4 q[2]; };
union FragC  { f32x8  v; float f[8]; };

// --------------------------------- utility ---------------------------------

__global__ void zero_f32(float* __restrict__ p, int n) {
    int i = blockIdx.x * blockDim.x + threadIdx.x;
    if (i < n) p[i] = 0.0f;
}

__global__ void degree_kernel(const int* __restrict__ row, float* __restrict__ deg, int E) {
    int e = blockIdx.x * blockDim.x + threadIdx.x;
    if (e < E) atomicAdd(&deg[row[e]], 1.0f);
}

__global__ void dinv_kernel(float* __restrict__ deg, int n) {
    int i = blockIdx.x * blockDim.x + threadIdx.x;
    if (i < n) {
        float d = deg[i];
        deg[i] = (d > 0.0f) ? rsqrtf(fmaxf(d, 1.0f)) : 0.0f;
    }
}

// f32 -> bf16 (RNE), 8 elements per thread, b128 in / b128 out
__global__ void cvt_bf16_kernel(const float* __restrict__ in,
                                unsigned short* __restrict__ outb, int n8) {
    int i = blockIdx.x * blockDim.x + threadIdx.x;
    if (i >= n8) return;
    const float4* p = (const float4*)(in + (size_t)i * 8);
    float4 a = p[0], b = p[1];
    uint4 o;
    o.x = pk2(a.x, a.y);
    o.y = pk2(a.z, a.w);
    o.z = pk2(b.x, b.y);
    o.w = pk2(b.z, b.w);
    *(uint4*)(outb + (size_t)i * 8) = o;
}

// Repack a 128x128 row-major f32 weight matrix into WMMA B-fragment order:
//   packed[((j*4 + kk)*32 + lane)*16 + e] = bf16( W[(kk*32 + 8*(lane/16) + (e%8) + 16*(e/8))][j*16 + lane%16] )
__global__ void pack_w_kernel(const float* __restrict__ W, unsigned short* __restrict__ out) {
    int t = blockIdx.x * blockDim.x + threadIdx.x;      // 16384 threads
    int e    = t & 15;
    int lane = (t >> 4) & 31;
    int kk   = (t >> 9) & 3;
    int j    = t >> 11;
    int h = lane >> 4, n = lane & 15;
    int k = kk * 32 + 8 * h + (e & 7) + 16 * (e >> 3);
    out[t] = f2bf(W[k * CCH + j * 16 + n]);
}

// ------------------------------ aggregation --------------------------------

// z[row] += -(dinv[row]*dinv[col]) * x[col]   (one thread per edge x 4 channels)
__global__ void aggregate_f32_kernel(const float* __restrict__ X, const int* __restrict__ ei,
                                     const float* __restrict__ dinv, float* __restrict__ Z, int E) {
    int gid = blockIdx.x * blockDim.x + threadIdx.x;
    int e = gid >> 5;
    if (e >= E) return;
    int c4 = (gid & 31) << 2;
    int r = ei[e];           // edge_index[0][e]
    int c = ei[E + e];       // edge_index[1][e]
    float w = -(dinv[r] * dinv[c]);
    const float4 xv = *(const float4*)(X + (size_t)c * CCH + c4);
    float* zp = Z + (size_t)r * CCH + c4;
    atomicAdd(zp + 0, w * xv.x);
    atomicAdd(zp + 1, w * xv.y);
    atomicAdd(zp + 2, w * xv.z);
    atomicAdd(zp + 3, w * xv.w);
}

// same, gathering from a bf16 feature matrix (halves gather bytes)
__global__ void aggregate_bf16_kernel(const unsigned short* __restrict__ Xb,
                                      const int* __restrict__ ei,
                                      const float* __restrict__ dinv,
                                      float* __restrict__ Z, int E) {
    int gid = blockIdx.x * blockDim.x + threadIdx.x;
    int e = gid >> 5;
    if (e >= E) return;
    int c4 = (gid & 31) << 2;
    int r = ei[e];
    int c = ei[E + e];
    float w = -(dinv[r] * dinv[c]);
    const uint2 xv = *(const uint2*)(Xb + (size_t)c * CCH + c4);
    float x0 = __uint_as_float(xv.x << 16);
    float x1 = __uint_as_float(xv.x & 0xffff0000u);
    float x2 = __uint_as_float(xv.y << 16);
    float x3 = __uint_as_float(xv.y & 0xffff0000u);
    float* zp = Z + (size_t)r * CCH + c4;
    atomicAdd(zp + 0, w * x0);
    atomicAdd(zp + 1, w * x1);
    atomicAdd(zp + 2, w * x2);
    atomicAdd(zp + 3, w * x3);
}

// ------------------------------ WMMA GEMM ----------------------------------

__device__ __forceinline__ bf16x16 load_frag16(const unsigned short* __restrict__ p0,
                                               const unsigned short* __restrict__ p1) {
    Frag16 u;
    u.q[0] = *(const u32x4*)p0;
    u.q[1] = *(const u32x4*)p1;
    return u.v;
}

// Y[16 rows x 128 cols per block] = epilogue( A1 @ W1 + A2 @ W2 + bias )
// 8 waves per block (one 16x16 WMMA tile per wave, wave id = column tile).
// mode 0: outb[idx] = bf16(relu(acc))          (layer-1)
// mode 1: outf[idx] = 0.5*(resid + relu(acc))  (layer-2 + residual)
__global__ __launch_bounds__(256) void gemm_wmma_kernel(
    const unsigned short* __restrict__ A1, const unsigned short* __restrict__ A2,
    const unsigned short* __restrict__ Bp1, const unsigned short* __restrict__ Bp2,
    const float* __restrict__ bias, const float* __restrict__ resid,
    float* __restrict__ outf, unsigned short* __restrict__ outb, int mode) {
    int wave = threadIdx.x >> 5;     // column tile j: 0..7
    int lane = threadIdx.x & 31;
    int h  = lane >> 4;
    int ln = lane & 15;              // A row within tile AND C/D column within tile
    int rowBase = blockIdx.x << 4;

    FragC acc;
    float bn = bias[wave * 16 + ln];
#pragma unroll
    for (int v = 0; v < 8; ++v) acc.f[v] = bn;   // bias depends on column only

    // 16-bit A-fragment: elements 0..7 <- k = 8h+0..7 ; 8..15 <- k = 16+8h+0..7
    const unsigned short* a1row = A1 + (size_t)(rowBase + ln) * CCH + 8 * h;
    const unsigned short* a2row = A2 + (size_t)(rowBase + ln) * CCH + 8 * h;
    const unsigned short* b1p = Bp1 + ((size_t)(wave * 4 * 32) + lane) * 16;
    const unsigned short* b2p = Bp2 + ((size_t)(wave * 4 * 32) + lane) * 16;

#pragma unroll
    for (int kk = 0; kk < 4; ++kk) {
        bf16x16 af1 = load_frag16(a1row + kk * 32, a1row + kk * 32 + 16);
        bf16x16 bf1 = load_frag16(b1p + kk * 512, b1p + kk * 512 + 8);
        acc.v = __builtin_amdgcn_wmma_f32_16x16x32_bf16(
            false, af1, false, bf1, (short)0, acc.v, false, false);
        bf16x16 af2 = load_frag16(a2row + kk * 32, a2row + kk * 32 + 16);
        bf16x16 bf2 = load_frag16(b2p + kk * 512, b2p + kk * 512 + 8);
        acc.v = __builtin_amdgcn_wmma_f32_16x16x32_bf16(
            false, af2, false, bf2, (short)0, acc.v, false, false);
    }

    // C/D layout: VGPR v, lanes 0-15 -> M=v, lanes 16-31 -> M=v+8, N=lane&15
#pragma unroll
    for (int v = 0; v < 8; ++v) {
        int m = rowBase + v + 8 * h;
        float y = fmaxf(acc.f[v], 0.0f);
        size_t idx = (size_t)m * CCH + wave * 16 + ln;
        if (mode) {
            outf[idx] = 0.5f * (resid[idx] + y);
        } else {
            outb[idx] = f2bf(y);
        }
    }
}

// ------------------------------- launcher ----------------------------------

extern "C" void kernel_launch(void* const* d_in, const int* in_sizes, int n_in,
                              void* d_out, int out_size, void* d_ws, size_t ws_size,
                              hipStream_t stream) {
    const float* x   = (const float*)d_in[0];
    const int*   ei  = (const int*)  d_in[1];
    const float* W00 = (const float*)d_in[2];
    const float* W01 = (const float*)d_in[3];
    const float* b0  = (const float*)d_in[4];
    const float* W10 = (const float*)d_in[5];
    const float* W11 = (const float*)d_in[6];
    const float* b1  = (const float*)d_in[7];
    float* out = (float*)d_out;

    const int C = CCH;
    const int N = in_sizes[0] / C;      // 40000
    const int E = in_sizes[1] / 2;      // 640000 directed edges
    const int NC = N * C;

    // workspace layout (256B-aligned sections)
    char* ws = (char*)d_ws;
    size_t off = 0;
    float* deg          = (float*)(ws + off);          off += (((size_t)N * 4) + 255) & ~(size_t)255;
    float* z            = (float*)(ws + off);          off += (size_t)NC * 4;
    unsigned short* xb  = (unsigned short*)(ws + off); off += (size_t)NC * 2;
    unsigned short* zb  = (unsigned short*)(ws + off); off += (size_t)NC * 2;
    unsigned short* y0b = (unsigned short*)(ws + off); off += (size_t)NC * 2;
    unsigned short* w00p = (unsigned short*)(ws + off);
    unsigned short* w01p = w00p + 16384;
    unsigned short* w10p = w00p + 2 * 16384;
    unsigned short* w11p = w00p + 3 * 16384;

    dim3 blk(256);
    const int n8 = NC / 8;

    // degree -> dinv (stored in-place in deg)
    zero_f32<<<(N + 255) / 256, blk, 0, stream>>>(deg, N);
    degree_kernel<<<(E + 255) / 256, blk, 0, stream>>>(ei, deg, E);
    dinv_kernel<<<(N + 255) / 256, blk, 0, stream>>>(deg, N);

    // one-time conversions / repacks
    cvt_bf16_kernel<<<(n8 + 255) / 256, blk, 0, stream>>>(x, xb, n8);
    pack_w_kernel<<<64, blk, 0, stream>>>(W00, w00p);
    pack_w_kernel<<<64, blk, 0, stream>>>(W01, w01p);
    pack_w_kernel<<<64, blk, 0, stream>>>(W10, w10p);
    pack_w_kernel<<<64, blk, 0, stream>>>(W11, w11p);

    // layer 1: z = L~ x (exact f32 gather) ; y0b = bf16(relu(x@W00 + z@W01 + b0))
    zero_f32<<<(NC + 255) / 256, blk, 0, stream>>>(z, NC);
    aggregate_f32_kernel<<<(E * 32 + 255) / 256, blk, 0, stream>>>(x, ei, deg, z, E);
    cvt_bf16_kernel<<<(n8 + 255) / 256, blk, 0, stream>>>(z, zb, n8);
    gemm_wmma_kernel<<<N / 16, blk, 0, stream>>>(xb, zb, w00p, w01p, b0, x, out, y0b, 0);

    // layer 2 + residual: z = L~ y0 (bf16 gather) ; out = 0.5*(x + relu(y0@W10 + z@W11 + b1))
    zero_f32<<<(NC + 255) / 256, blk, 0, stream>>>(z, NC);
    aggregate_bf16_kernel<<<(E * 32 + 255) / 256, blk, 0, stream>>>(y0b, ei, deg, z, E);
    cvt_bf16_kernel<<<(n8 + 255) / 256, blk, 0, stream>>>(z, zb, n8);
    gemm_wmma_kernel<<<N / 16, blk, 0, stream>>>(y0b, zb, w10p, w11p, b1, x, out, y0b, 1);
}
